// Block_spe_72464688218579
// MI455X (gfx1250) — compile-verified, matrix operand order checked by reading, not run on hardware
//
#include <hip/hip_runtime.h>
#include <math.h>

// Problem constants (match reference)
#define Bc 2
#define Sc 2048
#define Dc 1024
#define Hc 16
#define HDc 64
#define FFc 4096
#define DEPTHc 4
#define Mrows (Bc * Sc) // 4096 flattened rows

typedef __attribute__((ext_vector_type(16))) __bf16 v16bf;
typedef __attribute__((ext_vector_type(8)))  float  v8f;
typedef __attribute__((ext_vector_type(4)))  unsigned v4u;
typedef __attribute__((ext_vector_type(8)))  int      v8i;
typedef __attribute__((ext_vector_type(4)))  int      v4i;

union FragA { v16bf v; uint4 q[2]; };
union FragC { v8f v; float f[8]; };

#define WMMA_BF16(a, b, c) \
  __builtin_amdgcn_wmma_f32_16x16x32_bf16(false, (a), false, (b), (short)0, (c), false, false)

#if __has_builtin(__builtin_amdgcn_tensor_load_to_lds)
#define HAVE_TDM 1
#else
#define HAVE_TDM 0
#endif

// ---- fragment loads ----------------------------------------------------
// 16-bit A/B fragment layout (ISA 7.12.2): lanes 0-15 hold row r, k={0..7,16..23};
// lanes 16-31 hold row r, k={8..15,24..31}. Same pattern for A rows and B columns
// when B is stored transposed [N,K] row-major.
__device__ inline v16bf lds_frag(const __bf16* tile, int row0, int ld) {
  int lane = threadIdx.x & 31;
  const __bf16* p = tile + (size_t)(row0 + (lane & 15)) * ld + (lane >> 4) * 8;
  FragA f;
  f.q[0] = *(const uint4*)p;
  f.q[1] = *(const uint4*)(p + 16);
  return f.v;
}

__device__ inline v16bf g_frag(const __bf16* base, size_t ld) {
  int lane = threadIdx.x & 31;
  const __bf16* p = base + (size_t)(lane & 15) * ld + (lane >> 4) * 8;
  FragA f;
  f.q[0] = *(const uint4*)p;
  f.q[1] = *(const uint4*)(p + 16);
  return f.v;
}

#if HAVE_TDM
// ---- CDNA5 Tensor Data Mover: 2D tile (128 rows x 32 bf16) -> LDS -------
// D# per CDNA5 ISA.  LDS padding: +16B after every 64B
// row (pad_interval=3 -> 16 DWORDs, pad_amount=3 -> 4 DWORDs), matching the
// As/Bs row stride of 40 bf16.  This toolchain's builtin takes 6 args:
// (v4u group0, v8i group1, v4i group2, v4i group3, v8i extra, i32 cpol).
__device__ inline void tdm_load_tile(unsigned lds_off, const __bf16* gaddr, int K_elems) {
  unsigned long long ga = (unsigned long long)(uintptr_t)gaddr;
  v4u g0;
  g0[0] = 1u;                                                   // count=1 (valid), user mode
  g0[1] = lds_off;                                              // lds_addr
  g0[2] = (unsigned)ga;                                         // global_addr[31:0]
  g0[3] = (unsigned)((ga >> 32) & 0x01FFFFFFu) | (2u << 30);    // addr[56:32] | type=2
  v8i g1;
  g1[0] = (int)((1u << 16)      // data_size = 2 bytes
              | (1u << 20)      // pad_enable
              | (3u << 22)      // pad_interval: 16 DWORDs (64B)
              | (3u << 25));    // pad_amount: 4 DWORDs (16B)
  g1[1] = (int)(((unsigned)K_elems & 0xFFFFu) << 16);           // tensor_dim0[15:0]
  g1[2] = (int)(((unsigned)K_elems >> 16) | (0x4000u << 16));   // td0[31:16] | tensor_dim1 lo (16384)
  g1[3] = (int)(32u << 16);                                     // tensor_dim1 hi=0 | tile_dim0=32
  g1[4] = (int)128u;                                            // tile_dim1=128 | tile_dim2=0
  g1[5] = (int)K_elems;                                         // tensor_dim0_stride[31:0]
  g1[6] = 0;                                                    // stride hi / dim1_stride lo
  g1[7] = 0;
  v4i z4 = {0, 0, 0, 0};
  v8i z8 = {0, 0, 0, 0, 0, 0, 0, 0};
  __builtin_amdgcn_tensor_load_to_lds(g0, g1, z4, z4, z8, 0);
}
#else
// ---- fallback: CDNA5 async global->LDS copy (16B per lane), ASYNCcnt ----
__device__ inline void async_cp16(unsigned lds_off, const __bf16* g) {
  asm volatile("global_load_async_to_lds_b128 %0, %1, off"
               :: "v"(lds_off), "v"(g) : "memory");
}
#define WAIT_ASYNC_HAVE1 asm volatile("s_wait_asynccnt 0x4" ::: "memory")
#define WAIT_ASYNC_ALL   asm volatile("s_wait_asynccnt 0x0" ::: "memory")
#endif

// ---- elementwise converts ----------------------------------------------
__global__ void cvt_bf16_k(const float* __restrict__ s, __bf16* __restrict__ d, size_t n) {
  size_t i = (size_t)blockIdx.x * blockDim.x + threadIdx.x;
  if (i < n) d[i] = (__bf16)s[i];
}

// dst[n*K + k] = src[k*N + n]  (fp32 [K,N] -> bf16 [N,K]), LDS-tiled so both
// global streams are coalesced. K,N are multiples of 32 here.
__global__ __launch_bounds__(256) void tconv_k(const float* __restrict__ s,
                                               __bf16* __restrict__ d, int K, int N) {
  __shared__ float t[32][33];
  const int n0 = blockIdx.x * 32, k0 = blockIdx.y * 32;
  const int tx = threadIdx.x & 31, ty = threadIdx.x >> 5; // 32 x 8
  for (int i = ty; i < 32; i += 8)
    t[i][tx] = s[(size_t)(k0 + i) * N + n0 + tx];
  __syncthreads();
  for (int i = ty; i < 32; i += 8)
    d[(size_t)(n0 + i) * K + k0 + tx] = (__bf16)t[tx][i];
}

// ---- GEMM: C[M,N] = A[M,K](bf16) x Bt[N,K](bf16), fp32 accum, fused epilogue
// Double-buffered LDS tiles filled by the Tensor Data Mover (TENSOR_LOAD_TO_LDS,
// TENSORcnt) or, without the builtin, GLOBAL_LOAD_ASYNC_TO_LDS_B128 (ASYNCcnt).
#define BM 128
#define BN 128
#define BK 32
#define LDT 40 // padded LDS row stride (bf16 elems), 80B = 16B-aligned rows

// MODE 0: Q proj  -> (acc+b)*scale -> bf16 [B,H,S,HD]
// MODE 1: K proj  -> (acc+b)       -> bf16 [B,H,S,HD]
// MODE 2: V proj  -> (acc+b)       -> bf16 [B,H,HD,S] (transposed)
// MODE 3: MLP1    -> gelu(acc+b)   -> bf16 [M,N]
// MODE 4: MLP2    -> acc+b+o+x     -> fp32 x (in place) + bf16 x
template <int MODE>
__global__ __launch_bounds__(256) void gemm_bf16_k(
    const __bf16* __restrict__ A, const __bf16* __restrict__ Bw,
    const float* __restrict__ bias, const float* __restrict__ resO,
    float* xF, __bf16* outB, int N, int K, float scale) {
  __shared__ __bf16 As[2][BM][LDT];
  __shared__ __bf16 Bs[2][BN][LDT];

  const int tid = threadIdx.x;
  const int wid = tid >> 5;
  const int wm = wid >> 1, wn = wid & 1;   // 4x2 wave grid; wave tile 32x64
  const int bm0 = blockIdx.y * BM;
  const int bn0 = blockIdx.x * BN;

  FragC acc[2][4] = {};

#if HAVE_TDM
  const __bf16* ag0 = A + (size_t)bm0 * K;    // tile row 0
  const __bf16* bg0 = Bw + (size_t)bn0 * K;
  unsigned lA[2], lB[2];
  lA[0] = (unsigned)(uintptr_t)&As[0][0][0];
  lA[1] = (unsigned)(uintptr_t)&As[1][0][0];
  lB[0] = (unsigned)(uintptr_t)&Bs[0][0][0];
  lB[1] = (unsigned)(uintptr_t)&Bs[1][0][0];
  auto issue = [&](int step, int buf) {
    if (wid == 0) {
      tdm_load_tile(lA[buf], ag0 + (size_t)step * BK, K);
      tdm_load_tile(lB[buf], bg0 + (size_t)step * BK, K);
    }
  };
#else
  // staging: 256 threads, each moves 2x16B of A and 2x16B of B per tile
  const int trow = tid >> 1;            // 0..127
  const int tcol = (tid & 1) << 4;      // 0 or 16 (bf16 elems)
  const __bf16* ag = A + (size_t)(bm0 + trow) * K + tcol;
  const __bf16* bg = Bw + (size_t)(bn0 + trow) * K + tcol;
  unsigned lA[2], lB[2];
  lA[0] = (unsigned)(uintptr_t)&As[0][trow][tcol];
  lA[1] = (unsigned)(uintptr_t)&As[1][trow][tcol];
  lB[0] = (unsigned)(uintptr_t)&Bs[0][trow][tcol];
  lB[1] = (unsigned)(uintptr_t)&Bs[1][trow][tcol];
  auto issue = [&](int step, int buf) {
    const __bf16* a = ag + (size_t)step * BK;
    const __bf16* b = bg + (size_t)step * BK;
    async_cp16(lA[buf],      a);
    async_cp16(lA[buf] + 16, a + 8);
    async_cp16(lB[buf],      b);
    async_cp16(lB[buf] + 16, b + 8);
  };
#endif

  const int steps = K >> 5;
  issue(0, 0);
  for (int i = 0; i < steps; ++i) {
    const int cur = i & 1;
#if HAVE_TDM
    if (i + 1 < steps) {
      issue(i + 1, cur ^ 1);
      if (wid == 0) __builtin_amdgcn_s_wait_tensorcnt(2); // tile i landed
    } else {
      if (wid == 0) __builtin_amdgcn_s_wait_tensorcnt(0);
    }
#else
    if (i + 1 < steps) {
      issue(i + 1, cur ^ 1);
      WAIT_ASYNC_HAVE1;   // tile i landed; tile i+1 still in flight
    } else {
      WAIT_ASYNC_ALL;
    }
#endif
    __syncthreads();

    const __bf16* Ab = &As[cur][0][0];
    const __bf16* Bb = &Bs[cur][0][0];
    v16bf fa[2], fb[4];
    for (int mi = 0; mi < 2; ++mi) fa[mi] = lds_frag(Ab, wm * 32 + mi * 16, LDT);
    for (int ni = 0; ni < 4; ++ni) fb[ni] = lds_frag(Bb, wn * 64 + ni * 16, LDT);
    for (int mi = 0; mi < 2; ++mi)
      for (int ni = 0; ni < 4; ++ni)
        acc[mi][ni].v = WMMA_BF16(fa[mi], fb[ni], acc[mi][ni].v);
    __syncthreads();
  }

  // epilogue: C element (m,n): vgpr r, lane = (n%16) + 16*((m%16)/8)
  const int lane = tid & 31;
  const int hl = lane >> 4, c16 = lane & 15;
  for (int mi = 0; mi < 2; ++mi) {
    for (int ni = 0; ni < 4; ++ni) {
      const int gn = bn0 + wn * 64 + ni * 16 + c16;
      const float bv = bias[gn];
      for (int r = 0; r < 8; ++r) {
        const int gm = bm0 + wm * 32 + mi * 16 + hl * 8 + r;
        float v = acc[mi][ni].f[r];
        if constexpr (MODE <= 2) {
          v += bv;
          if constexpr (MODE == 0) v *= scale;
          const int bb = gm >> 11;          // / Sc
          const int ss = gm & (Sc - 1);
          const int hh = gn >> 6;           // / HDc
          const int hd = gn & (HDc - 1);
          if constexpr (MODE == 2)
            outB[(((size_t)bb * Hc + hh) * HDc + hd) * Sc + ss] = (__bf16)v;
          else
            outB[(((size_t)bb * Hc + hh) * Sc + ss) * HDc + hd] = (__bf16)v;
        } else if constexpr (MODE == 3) {
          v += bv;
          const float g = 0.5f * v * (1.0f + erff(v * 0.70710678118654752f));
          outB[(size_t)gm * N + gn] = (__bf16)g;
        } else {
          const size_t idx = (size_t)gm * N + gn;
          v += bv + resO[idx] + xF[idx];
          xF[idx] = v;
          outB[idx] = (__bf16)v;
        }
      }
    }
  }
}

// ---- flash attention: scores = Q@K^T (scale pre-applied to Q), online softmax,
// O = P@V.  4 waves/block, each wave owns 32 q rows; loop over t in 32-col tiles.
__global__ __launch_bounds__(128) void attn_k(
    const __bf16* __restrict__ qb, const __bf16* __restrict__ kb,
    const __bf16* __restrict__ vt, const int* __restrict__ mask,
    float* __restrict__ oF, __bf16* __restrict__ oB) {
  __shared__ __bf16 Pl[4][32][LDT];

  const int bh = blockIdx.x;             // b*H + h
  const int b = bh >> 4, h = bh & 15;
  const int wid = threadIdx.x >> 5;
  const int lane = threadIdx.x & 31;
  const int hl = lane >> 4, c16 = lane & 15;
  const int q0 = blockIdx.y * 128 + wid * 32;

  const __bf16* Qb = qb + ((size_t)bh * Sc + q0) * HDc;
  const __bf16* Kb = kb + (size_t)bh * Sc * HDc;
  const __bf16* Vb = vt + (size_t)bh * HDc * Sc;
  const int* mrow = mask + (size_t)b * (3 * Sc); // modality 'a' slice

  v16bf fq[2][2];
  for (int mi = 0; mi < 2; ++mi)
    for (int kk = 0; kk < 2; ++kk)
      fq[mi][kk] = g_frag(Qb + (size_t)(mi * 16) * HDc + kk * 32, HDc);

  FragC accO[2][4] = {};
  float rowm[2][8], rowl[2][8];
  for (int mi = 0; mi < 2; ++mi)
    for (int r = 0; r < 8; ++r) { rowm[mi][r] = -INFINITY; rowl[mi][r] = 0.0f; }

  for (int t0 = 0; t0 < Sc; t0 += 32) {
    FragC sc[2][2] = {};
    v16bf fkA[2], fkB[2];
    for (int ni = 0; ni < 2; ++ni) {
      fkA[ni] = g_frag(Kb + (size_t)(t0 + ni * 16) * HDc, HDc);
      fkB[ni] = g_frag(Kb + (size_t)(t0 + ni * 16) * HDc + 32, HDc);
    }
    for (int mi = 0; mi < 2; ++mi)
      for (int ni = 0; ni < 2; ++ni) {
        sc[mi][ni].v = WMMA_BF16(fq[mi][0], fkA[ni], sc[mi][ni].v);
        sc[mi][ni].v = WMMA_BF16(fq[mi][1], fkB[ni], sc[mi][ni].v);
      }

    const int mk0 = mrow[t0 + c16];
    const int mk1 = mrow[t0 + 16 + c16];

    for (int mi = 0; mi < 2; ++mi) {
      for (int r = 0; r < 8; ++r) {
        float s0 = mk0 ? sc[mi][0].f[r] : -INFINITY;
        float s1 = mk1 ? sc[mi][1].f[r] : -INFINITY;
        float mx = fmaxf(s0, s1);
        mx = fmaxf(mx, __shfl_xor(mx, 1, 32));
        mx = fmaxf(mx, __shfl_xor(mx, 2, 32));
        mx = fmaxf(mx, __shfl_xor(mx, 4, 32));
        mx = fmaxf(mx, __shfl_xor(mx, 8, 32));
        const float mo = rowm[mi][r];
        const float mn = fmaxf(mo, mx);
        const float c = (mo == mn) ? 1.0f : __expf(mo - mn);
        const bool dead = (mn == -INFINITY); // fully-masked row so far
        const float p0 = dead ? 0.0f : __expf(s0 - mn);
        const float p1 = dead ? 0.0f : __expf(s1 - mn);
        float rs = p0 + p1;
        rs += __shfl_xor(rs, 1, 32);
        rs += __shfl_xor(rs, 2, 32);
        rs += __shfl_xor(rs, 4, 32);
        rs += __shfl_xor(rs, 8, 32);
        rowl[mi][r] = rowl[mi][r] * c + rs;
        rowm[mi][r] = mn;
        for (int nj = 0; nj < 4; ++nj) accO[mi][nj].f[r] *= c;
        Pl[wid][mi * 16 + hl * 8 + r][c16]      = (__bf16)p0;
        Pl[wid][mi * 16 + hl * 8 + r][16 + c16] = (__bf16)p1;
      }
    }
    // same-wave LDS RAW: DS ops are in order, but be explicit
    asm volatile("s_wait_dscnt 0" ::: "memory");

    v16bf fp0 = lds_frag(&Pl[wid][0][0], 0, LDT);
    v16bf fp1 = lds_frag(&Pl[wid][0][0], 16, LDT);
    for (int nj = 0; nj < 4; ++nj) {
      v16bf fv = g_frag(Vb + (size_t)(nj * 16) * Sc + t0, Sc);
      accO[0][nj].v = WMMA_BF16(fp0, fv, accO[0][nj].v);
      accO[1][nj].v = WMMA_BF16(fp1, fv, accO[1][nj].v);
    }
  }

  for (int mi = 0; mi < 2; ++mi) {
    for (int r = 0; r < 8; ++r) {
      const float l = rowl[mi][r];
      const float inv = (l > 0.0f) ? 1.0f / l : 0.0f; // masked rows -> 0
      const int ss = q0 + mi * 16 + hl * 8 + r;
      const size_t row = (size_t)b * Sc + ss;
      for (int nj = 0; nj < 4; ++nj) {
        const int col = h * HDc + nj * 16 + c16;
        const float v = accO[mi][nj].f[r] * inv;
        oF[row * Dc + col] = v;
        oB[row * Dc + col] = (__bf16)v;
      }
    }
  }
}

// ---- host-side orchestration -------------------------------------------
extern "C" void kernel_launch(void* const* d_in, const int* in_sizes, int n_in,
                              void* d_out, int out_size, void* d_ws, size_t ws_size,
                              hipStream_t stream) {
  const float* x  = (const float*)d_in[0];
  const int*   mk = (const int*)d_in[1];
  const float* Wq = (const float*)d_in[2];
  const float* bq = (const float*)d_in[3];
  const float* Wk = (const float*)d_in[4];
  const float* bk = (const float*)d_in[5];
  const float* Wv = (const float*)d_in[6];
  const float* bv = (const float*)d_in[7];
  const float* W1 = (const float*)d_in[8];
  const float* b1 = (const float*)d_in[9];
  const float* W2 = (const float*)d_in[10];
  const float* b2 = (const float*)d_in[11];
  float* xF = (float*)d_out; // running fp32 activations, updated in place

  char* wp = (char*)d_ws;
  auto carve = [&](size_t bytes) {
    void* p = (void*)wp;
    wp += (bytes + 255) & ~(size_t)255;
    return p;
  };
  __bf16* xB  = (__bf16*)carve((size_t)Mrows * Dc * 2);
  __bf16* qB  = (__bf16*)carve((size_t)Mrows * Dc * 2);
  __bf16* kB  = (__bf16*)carve((size_t)Mrows * Dc * 2);
  __bf16* vtB = (__bf16*)carve((size_t)Mrows * Dc * 2);
  __bf16* oB  = (__bf16*)carve((size_t)Mrows * Dc * 2);
  float*  oF  = (float*)carve((size_t)Mrows * Dc * 4);
  __bf16* hB  = (__bf16*)carve((size_t)Mrows * FFc * 2);
  __bf16* wtq = (__bf16*)carve((size_t)Dc * Dc * 2);
  __bf16* wtk = (__bf16*)carve((size_t)Dc * Dc * 2);
  __bf16* wtv = (__bf16*)carve((size_t)Dc * Dc * 2);
  __bf16* wt1 = (__bf16*)carve((size_t)Dc * FFc * 2);
  __bf16* wt2 = (__bf16*)carve((size_t)FFc * Dc * 2);

  const float scale = 0.125f; // HD^-0.5

  // x -> d_out (running fp32) and bf16 copy
  (void)hipMemcpyAsync(xF, x, (size_t)Mrows * Dc * 4, hipMemcpyDeviceToDevice, stream);
  {
    size_t n = (size_t)Mrows * Dc;
    cvt_bf16_k<<<dim3((unsigned)((n + 255) / 256)), 256, 0, stream>>>(x, xB, n);
  }

  const dim3 gD(Dc / BN, Mrows / BM);   // N=1024 GEMMs
  const dim3 gF(FFc / BN, Mrows / BM);  // N=4096 GEMM
  const dim3 gAttn(Bc * Hc, Sc / 128);

  for (int l = 0; l < DEPTHc; ++l) {
    // weights -> bf16 [N,K] (tiled transpose, coalesced both ways)
    tconv_k<<<dim3(Dc / 32, Dc / 32), 256, 0, stream>>>(Wq + (size_t)l * Dc * Dc, wtq, Dc, Dc);
    tconv_k<<<dim3(Dc / 32, Dc / 32), 256, 0, stream>>>(Wk + (size_t)l * Dc * Dc, wtk, Dc, Dc);
    tconv_k<<<dim3(Dc / 32, Dc / 32), 256, 0, stream>>>(Wv + (size_t)l * Dc * Dc, wtv, Dc, Dc);
    tconv_k<<<dim3(FFc / 32, Dc / 32), 256, 0, stream>>>(W1 + (size_t)l * Dc * FFc, wt1, Dc, FFc);
    tconv_k<<<dim3(Dc / 32, FFc / 32), 256, 0, stream>>>(W2 + (size_t)l * FFc * Dc, wt2, FFc, Dc);
    // QKV projections
    gemm_bf16_k<0><<<gD, 256, 0, stream>>>(xB, wtq, bq + (size_t)l * Dc, nullptr, nullptr, qB,  Dc, Dc, scale);
    gemm_bf16_k<1><<<gD, 256, 0, stream>>>(xB, wtk, bk + (size_t)l * Dc, nullptr, nullptr, kB,  Dc, Dc, 1.0f);
    gemm_bf16_k<2><<<gD, 256, 0, stream>>>(xB, wtv, bv + (size_t)l * Dc, nullptr, nullptr, vtB, Dc, Dc, 1.0f);
    // attention
    attn_k<<<gAttn, 128, 0, stream>>>(qB, kB, vtB, mk, oF, oB);
    // MLP
    gemm_bf16_k<3><<<gF, 256, 0, stream>>>(oB, wt1, b1 + (size_t)l * FFc, nullptr, nullptr, hB, FFc, Dc, 1.0f);
    gemm_bf16_k<4><<<gD, 256, 0, stream>>>(hB, wt2, b2 + (size_t)l * Dc, oF, xF, xB, Dc, FFc, 1.0f);
  }
}
